// DARNN_77214922047720
// MI455X (gfx1250) — compile-verified, matrix operand order checked by reading, not run on hardware
//
#include <hip/hip_runtime.h>
#include <hip/hip_bf16.h>

// ---------------- CDNA5 WMMA types & helpers (wave32) ----------------
typedef __attribute__((ext_vector_type(16))) _Float16 v16h;
typedef __attribute__((ext_vector_type(8)))  float    v8f;

__device__ __forceinline__ float fast_tanh(float x) {
#if __has_builtin(__builtin_amdgcn_tanhf)
  return __builtin_amdgcn_tanhf(x);
#else
  float r; asm volatile("v_tanh_f32 %0, %1" : "=v"(r) : "v"(x)); return r;
#endif
}
__device__ __forceinline__ float fast_sigmoid(float x) {
  return 0.5f * fast_tanh(0.5f * x) + 0.5f;
}
__device__ __forceinline__ float fast_exp(float x) {
#if __has_builtin(__builtin_amdgcn_exp2f)
  return __builtin_amdgcn_exp2f(x * 1.4426950408889634f);
#else
  return __expf(x);
#endif
}

// ---- fragment-native ("swizzled") storage -----------------------------
// A WMMA operand fragment is the per-lane register image: 32 lanes x 16
// halves, stored contiguously as [frag][lane][16]. One fragment = 512
// halves = 1KB. Loading it is one 32B vector load per lane
// (-> 2x b128 loads) instead of 16 strided u16 gathers.
//
// ISA 7.12.2 mapping used when building the image:
//   lane l: r = l&15 (row / col), kh = (l>>4)*8
//   reg i<8 : k = kh + i ; reg i>=8 : k = 16 + kh + (i-8) = 8 + kh + i
__device__ __forceinline__ v16h load_frag(const _Float16* base) {
  return *(const v16h*)(base + (size_t)(threadIdx.x & 31) * 16);
}

// swizzled flat index for logical (r, k) of an A matrix (16 x K)
__device__ __forceinline__ int a_sw_index(int r, int k) {
  int ks = k >> 5, kr = k & 31;
  int hi, i;
  if (kr < 16) { hi = kr >> 3; i = kr & 7; }
  else         { hi = (kr - 16) >> 3; i = 8 + (kr & 7); }
  return ks * 512 + (hi * 16 + r) * 16 + i;
}

// build one element of a swizzled B image for W^T where the logical
// element is (k, col) = src[col*K + k]  (src row-major (N x K))
__device__ __forceinline__ void swz_store(_Float16* dst, const float* src,
                                          long i, int K, int N) {
  int i16 = (int)(i & 15);
  int l   = (int)((i >> 4) & 31);
  long tile = i >> 9;
  int NT = N >> 4;
  int nt = (int)(tile % NT), ks = (int)(tile / NT);
  int col = nt * 16 + (l & 15);
  int kh  = (l >> 4) * 8;
  int k   = ks * 32 + (i16 < 8 ? kh + i16 : 8 + kh + i16);
  dst[i] = (_Float16)src[(long)col * K + k];
}

// A fragment 16x32 f16, element (r,k) = src[r*ld + k]  (plain layout)
__device__ __forceinline__ v16h load_a_rm_f16(const _Float16* src, int ld) {
  const int l = threadIdx.x & 31;
  const int r = l & 15;
  const int kh = (l >> 4) * 8;
  v16h a;
  const _Float16* p = src + r * ld + kh;
#pragma unroll
  for (int i = 0; i < 8; ++i) a[i] = p[i];
#pragma unroll
  for (int i = 0; i < 8; ++i) a[8 + i] = p[16 + i];
  return a;
}
// A fragment 16x32 from f32 source stored transposed: (r,k)=src[k*ld+r]
__device__ __forceinline__ v16h load_a_cm_f32(const float* src, int ld) {
  const int l = threadIdx.x & 31;
  const int r = l & 15;
  const int kh = (l >> 4) * 8;
  v16h a;
#pragma unroll
  for (int i = 0; i < 8; ++i) a[i] = (_Float16)src[(kh + i) * ld + r];
#pragma unroll
  for (int i = 0; i < 8; ++i) a[8 + i] = (_Float16)src[(16 + kh + i) * ld + r];
  return a;
}
__device__ __forceinline__ v8f wmma_f16(v16h a, v16h b, v8f c) {
  return __builtin_amdgcn_wmma_f32_16x16x32_f16(false, a, false, b, (short)0, c,
                                                false, false);
}

// ---------------- K0: weights -> fragment-native f16 images ------------
__global__ __launch_bounds__(256) void prep_kernel(
    const float* __restrict__ We, const float* __restrict__ Ue,
    const float* __restrict__ Wih_e, const float* __restrict__ Whh_e,
    const float* __restrict__ bih_e, const float* __restrict__ bhh_e,
    const float* __restrict__ Wd, const float* __restrict__ Ud,
    const float* __restrict__ Whh_d, const float* __restrict__ bih_d,
    const float* __restrict__ bhh_d, const float* __restrict__ W1,
    _Float16* WeSW, _Float16* UeSW, _Float16* WiheSW, _Float16* WhheSW,
    _Float16* UdSW, _Float16* WdSW, _Float16* WhhdSW, _Float16* W1SW,
    float* be, float* bd)
{
  long i = (long)blockIdx.x * 256 + threadIdx.x;
  if (i < 65536)  { swz_store(WeSW,   We,    i, 512, 128);  return; } i -= 65536;
  if (i < 16384)  { swz_store(UeSW,   Ue,    i, 128, 128);  return; } i -= 16384;
  if (i < 131072) { swz_store(WiheSW, Wih_e, i, 128, 1024); return; } i -= 131072;
  if (i < 262144) { swz_store(WhheSW, Whh_e, i, 256, 1024); return; } i -= 262144;
  if (i < 65536)  { swz_store(UdSW,   Ud,    i, 256, 256);  return; } i -= 65536;
  if (i < 131072) { swz_store(WdSW,   Wd,    i, 512, 256);  return; } i -= 131072;
  if (i < 262144) { swz_store(WhhdSW, Whh_d, i, 256, 1024); return; } i -= 262144;
  if (i < 131072) { swz_store(W1SW,   W1,    i, 512, 256);  return; } i -= 131072;
  if (i < 1024)   { be[i] = bih_e[i] + bhh_e[i];            return; } i -= 1024;
  if (i < 1024)   { bd[i] = bih_d[i] + bhh_d[i];            return; }
}

// ---------------- K1: Ux[b][n][t] = sum_s x[b][s][n] * Ue[t][s] --------
__global__ __launch_bounds__(256) void ux_kernel(
    const float* __restrict__ x, const _Float16* __restrict__ UeSW,
    _Float16* __restrict__ Ux)
{
  const int b = blockIdx.x;
  const int wave = threadIdx.x >> 5;
  const int n0 = wave * 16;
  const float* xb = x + (size_t)b * 128 * 128;
  const int l = threadIdx.x & 31, col = l & 15, r0 = (l >> 4) * 8;
  v16h afr[4];
#pragma unroll
  for (int ks = 0; ks < 4; ++ks)     // A(r=n_local, k=s) = x[b][s][n0+r]
    afr[ks] = load_a_cm_f32(xb + ks * 32 * 128 + n0, 128);
#pragma unroll
  for (int tt = 0; tt < 8; ++tt) {
    v8f acc = {};
#pragma unroll
    for (int ks = 0; ks < 4; ++ks)
      acc = wmma_f16(afr[ks], load_frag(UeSW + (size_t)(ks * 8 + tt) * 512), acc);
#pragma unroll
    for (int i = 0; i < 8; ++i)
      Ux[((size_t)b * 128 + n0 + r0 + i) * 128 + tt * 16 + col] = (_Float16)acc[i];
  }
}

// ---------------- K2: encoder (input attention + LSTM), 16 batches/WG --
__global__ __launch_bounds__(256) void enc_kernel(
    const _Float16* __restrict__ Ux, const _Float16* __restrict__ WeSW,
    const _Float16* __restrict__ WiheSW, const _Float16* __restrict__ WhheSW,
    const float* __restrict__ be, const float* __restrict__ Ve,
    _Float16* __restrict__ H)
{
  __shared__ float sh_h[16][256];
  __shared__ float sh_c[16][256];
  __shared__ alignas(32) _Float16 sh_hsA[16 * 512];   // swizzled A, K=512
  __shared__ alignas(32) _Float16 sh_xtA[4 * 512];    // swizzled A, K=128
  __shared__ float sh_ohs[16][128];
  __shared__ float sh_e[16][128];
  __shared__ float sh_z[16][1024];
  __shared__ float sh_Ve[128];
  __shared__ float sh_be[1024];

  const int tid = threadIdx.x;
  const int wave = tid >> 5;
  const int b0 = blockIdx.x * 16;
  const int l = tid & 31, col = l & 15, r0 = (l >> 4) * 8;

  for (int i = tid; i < 16 * 256; i += 256) { (&sh_h[0][0])[i] = 0.f; (&sh_c[0][0])[i] = 0.f; }
  for (int i = tid; i < 128;  i += 256) sh_Ve[i] = Ve[i];
  for (int i = tid; i < 1024; i += 256) sh_be[i] = be[i];
  __syncthreads();

  for (int t = 0; t < 128; ++t) {
    // hs = [h | s] written directly as swizzled A image
    for (int j = tid; j < 16 * 512; j += 256) {
      int i16 = j & 15, ln = (j >> 4) & 31, ks = j >> 9;
      int r = ln & 15, kh = (ln >> 4) * 8;
      int k = ks * 32 + (i16 < 8 ? kh + i16 : 8 + kh + i16);
      sh_hsA[j] = (_Float16)(k < 256 ? sh_h[r][k] : sh_c[r][k - 256]);
    }
    __syncthreads();

    // ohs = hs @ We^T : (16 x 128), wave -> column tile
    {
      v8f acc = {};
#pragma unroll
      for (int ks = 0; ks < 16; ++ks)
        acc = wmma_f16(load_frag(sh_hsA + ks * 512),
                       load_frag(WeSW + (size_t)(ks * 8 + wave) * 512), acc);
#pragma unroll
      for (int i = 0; i < 8; ++i) sh_ohs[r0 + i][wave * 16 + col] = acc[i];
    }
    __syncthreads();

    // e[b][n] = sum_t tanh(ohs[b][t] + Ux[b][n][t]) * Ve[t]
    for (int idx = tid; idx < 2048; idx += 256) {
      int b = idx >> 7, n = idx & 127;
      const _Float16* up = Ux + ((size_t)(b0 + b) * 128 + n) * 128;
      float acc = 0.f;
#pragma unroll 4
      for (int tt = 0; tt < 128; ++tt)
        acc += fast_tanh(sh_ohs[b][tt] + (float)up[tt]) * sh_Ve[tt];
      sh_e[b][n] = acc;
    }
    __syncthreads();

    // softmax over n; quirk: xt = alpha * e (written swizzled)
    if (tid < 16) {
      int b = tid;
      float m = -3.4e38f;
      for (int n = 0; n < 128; ++n) m = fmaxf(m, sh_e[b][n]);
      float den = 0.f;
      for (int n = 0; n < 128; ++n) den += fast_exp(sh_e[b][n] - m);
      float inv = 1.f / den;
      for (int n = 0; n < 128; ++n) {
        float al = fast_exp(sh_e[b][n] - m) * inv;
        sh_xtA[a_sw_index(b, n)] = (_Float16)(al * sh_e[b][n]);
      }
    }
    __syncthreads();

    // z = xt @ Wih_e^T + h @ Whh_e^T : (16 x 1024); wave -> 8 col tiles
    {
      v16h ax[4], ah[8];
#pragma unroll
      for (int ks = 0; ks < 4; ++ks) ax[ks] = load_frag(sh_xtA + ks * 512);
#pragma unroll
      for (int ks = 0; ks < 8; ++ks) ah[ks] = load_frag(sh_hsA + ks * 512);
#pragma unroll
      for (int j = 0; j < 8; ++j) {
        int nt = wave * 8 + j;
        int c0 = nt * 16;
        v8f acc = {};
#pragma unroll
        for (int ks = 0; ks < 4; ++ks)
          acc = wmma_f16(ax[ks], load_frag(WiheSW + (size_t)(ks * 64 + nt) * 512), acc);
#pragma unroll
        for (int ks = 0; ks < 8; ++ks)
          acc = wmma_f16(ah[ks], load_frag(WhheSW + (size_t)(ks * 64 + nt) * 512), acc);
#pragma unroll
        for (int i = 0; i < 8; ++i) sh_z[r0 + i][c0 + col] = acc[i];
      }
    }
    __syncthreads();

    // pointwise LSTM (i,f,g,o)
    for (int idx = tid; idx < 4096; idx += 256) {
      int b = idx >> 8, mm = idx & 255;
      float zi = sh_z[b][mm]        + sh_be[mm];
      float zf = sh_z[b][256 + mm]  + sh_be[256 + mm];
      float zg = sh_z[b][512 + mm]  + sh_be[512 + mm];
      float zo = sh_z[b][768 + mm]  + sh_be[768 + mm];
      float ig = fast_sigmoid(zi), fg = fast_sigmoid(zf);
      float gg = fast_tanh(zg),    og = fast_sigmoid(zo);
      float c2 = fg * sh_c[b][mm] + ig * gg;
      float hn = og * fast_tanh(c2);
      sh_c[b][mm] = c2;
      sh_h[b][mm] = hn;
      H[((size_t)(b0 + b) * 128 + t) * 256 + mm] = (_Float16)hn;
    }
    __syncthreads();
  }
}

// ---------------- K3: Udx[b][t][k] = sum_m H[b][t][m] * Ud[k][m] -------
__global__ __launch_bounds__(256) void udx_kernel(
    const _Float16* __restrict__ H, const _Float16* __restrict__ UdSW,
    _Float16* __restrict__ Udx)
{
  const int b = blockIdx.x;
  const int wave = threadIdx.x >> 5;
  const int t0 = wave * 16;
  const _Float16* Hb = H + (size_t)b * 128 * 256;
  const int l = threadIdx.x & 31, col = l & 15, r0 = (l >> 4) * 8;
  v16h ah[8];
#pragma unroll
  for (int ks = 0; ks < 8; ++ks)
    ah[ks] = load_a_rm_f16(Hb + (size_t)t0 * 256 + ks * 32, 256);
#pragma unroll
  for (int kt = 0; kt < 16; ++kt) {
    v8f acc = {};
#pragma unroll
    for (int ks = 0; ks < 8; ++ks)
      acc = wmma_f16(ah[ks], load_frag(UdSW + (size_t)(ks * 16 + kt) * 512), acc);
#pragma unroll
    for (int i = 0; i < 8; ++i)
      Udx[((size_t)b * 128 + t0 + r0 + i) * 256 + kt * 16 + col] = (_Float16)acc[i];
  }
}

// ---------------- K4: decoder (temporal attention + LSTM + head) -------
__global__ __launch_bounds__(256) void dec_kernel(
    const _Float16* __restrict__ Udx, const _Float16* __restrict__ H,
    const _Float16* __restrict__ WdSW, const _Float16* __restrict__ WhhdSW,
    const float* __restrict__ bd, const float* __restrict__ Vd,
    const float* __restrict__ Wih_d, const float* __restrict__ Wt,
    const float* __restrict__ bt, const float* __restrict__ y_known,
    const _Float16* __restrict__ W1SW, const float* __restrict__ b1,
    const float* __restrict__ W2, const float* __restrict__ b2,
    float* __restrict__ out)
{
  __shared__ float sh_d[16][256];
  __shared__ float sh_c[16][256];
  __shared__ alignas(32) _Float16 sh_dsA[16 * 512];   // swizzled A, K=512
  __shared__ float sh_ods[16][256];
  __shared__ float sh_beta[16][128];
  __shared__ float sh_ctx[16][256];
  __shared__ float sh_z[16][1024];
  __shared__ float sh_Vd[256];
  __shared__ float sh_bd[1024];
  __shared__ float sh_yt[16];

  const int tid = threadIdx.x;
  const int wave = tid >> 5;
  const int b0 = blockIdx.x * 16;
  const int l = tid & 31, col = l & 15, r0 = (l >> 4) * 8;

  for (int i = tid; i < 16 * 256; i += 256) {
    (&sh_d[0][0])[i] = 0.f; (&sh_c[0][0])[i] = 0.f; (&sh_ctx[0][0])[i] = 0.f;
  }
  for (int i = tid; i < 256;  i += 256) sh_Vd[i] = Vd[i];
  for (int i = tid; i < 1024; i += 256) sh_bd[i] = bd[i];
  __syncthreads();

  for (int t = 0; t < 128; ++t) {
    // ds = [d | s] swizzled
    for (int j = tid; j < 16 * 512; j += 256) {
      int i16 = j & 15, ln = (j >> 4) & 31, ks = j >> 9;
      int r = ln & 15, kh = (ln >> 4) * 8;
      int k = ks * 32 + (i16 < 8 ? kh + i16 : 8 + kh + i16);
      sh_dsA[j] = (_Float16)(k < 256 ? sh_d[r][k] : sh_c[r][k - 256]);
    }
    __syncthreads();

    // ods = ds @ Wd^T : (16 x 256); wave -> 2 col tiles
#pragma unroll
    for (int j = 0; j < 2; ++j) {
      int nt = wave * 2 + j;
      int c0 = nt * 16;
      v8f acc = {};
#pragma unroll
      for (int ks = 0; ks < 16; ++ks)
        acc = wmma_f16(load_frag(sh_dsA + ks * 512),
                       load_frag(WdSW + (size_t)(ks * 16 + nt) * 512), acc);
#pragma unroll
      for (int i = 0; i < 8; ++i) sh_ods[r0 + i][c0 + col] = acc[i];
    }
    __syncthreads();

    // l[b][t2] = sum_m tanh(ods[b][m] + Udx[b][t2][m]) * Vd[m]
    for (int idx = tid; idx < 2048; idx += 256) {
      int b = idx >> 7, t2 = idx & 127;
      const _Float16* up = Udx + ((size_t)(b0 + b) * 128 + t2) * 256;
      float acc = 0.f;
#pragma unroll 4
      for (int m = 0; m < 256; ++m)
        acc += fast_tanh(sh_ods[b][m] + (float)up[m]) * sh_Vd[m];
      sh_beta[b][t2] = acc;
    }
    __syncthreads();

    if (tid < 16) {
      int b = tid;
      float m = -3.4e38f;
      for (int i = 0; i < 128; ++i) m = fmaxf(m, sh_beta[b][i]);
      float den = 0.f;
      for (int i = 0; i < 128; ++i) den += fast_exp(sh_beta[b][i] - m);
      float inv = 1.f / den;
      for (int i = 0; i < 128; ++i) sh_beta[b][i] = fast_exp(sh_beta[b][i] - m) * inv;
    }
    __syncthreads();

    // context c[b][m] = sum_t2 beta * H[b][t2][m]
    for (int idx = tid; idx < 4096; idx += 256) {
      int b = idx >> 8, mm = idx & 255;
      const _Float16* hp = H + (size_t)(b0 + b) * 128 * 256 + mm;
      float acc = 0.f;
#pragma unroll 4
      for (int t2 = 0; t2 < 128; ++t2)
        acc += sh_beta[b][t2] * (float)hp[(size_t)t2 * 256];
      sh_ctx[b][mm] = acc;
    }
    __syncthreads();

    // y_tilde = [yt, c] @ Wt^T + bt
    if (tid < 16) {
      int b = tid;
      float acc = Wt[0] * y_known[(size_t)(b0 + b) * 128 + t] + bt[0];
      for (int mm = 0; mm < 256; ++mm) acc += Wt[1 + mm] * sh_ctx[b][mm];
      sh_yt[b] = acc;
    }
    __syncthreads();

    // z = d @ Whh_d^T (y_tilde * Wih_d added pointwise)
    {
      v16h ah[8];
#pragma unroll
      for (int ks = 0; ks < 8; ++ks) ah[ks] = load_frag(sh_dsA + ks * 512);
#pragma unroll
      for (int j = 0; j < 8; ++j) {
        int nt = wave * 8 + j;
        int c0 = nt * 16;
        v8f acc = {};
#pragma unroll
        for (int ks = 0; ks < 8; ++ks)
          acc = wmma_f16(ah[ks], load_frag(WhhdSW + (size_t)(ks * 64 + nt) * 512), acc);
#pragma unroll
        for (int i = 0; i < 8; ++i) sh_z[r0 + i][c0 + col] = acc[i];
      }
    }
    __syncthreads();

    for (int idx = tid; idx < 4096; idx += 256) {
      int b = idx >> 8, mm = idx & 255;
      float yt = sh_yt[b];
      float zi = sh_z[b][mm]        + yt * Wih_d[mm]        + sh_bd[mm];
      float zf = sh_z[b][256 + mm]  + yt * Wih_d[256 + mm]  + sh_bd[256 + mm];
      float zg = sh_z[b][512 + mm]  + yt * Wih_d[512 + mm]  + sh_bd[512 + mm];
      float zo = sh_z[b][768 + mm]  + yt * Wih_d[768 + mm]  + sh_bd[768 + mm];
      float ig = fast_sigmoid(zi), fg = fast_sigmoid(zf);
      float gg = fast_tanh(zg),    og = fast_sigmoid(zo);
      float c2 = fg * sh_c[b][mm] + ig * gg;
      sh_c[b][mm] = c2;
      sh_d[b][mm] = og * fast_tanh(c2);
    }
    __syncthreads();
  }

  // head: out1 = [d | c] @ W1^T + b1 ; out = out1 @ W2^T + b2
  for (int j = tid; j < 16 * 512; j += 256) {
    int i16 = j & 15, ln = (j >> 4) & 31, ks = j >> 9;
    int r = ln & 15, kh = (ln >> 4) * 8;
    int k = ks * 32 + (i16 < 8 ? kh + i16 : 8 + kh + i16);
    sh_dsA[j] = (_Float16)(k < 256 ? sh_d[r][k] : sh_ctx[r][k - 256]);
  }
  __syncthreads();
#pragma unroll
  for (int j = 0; j < 2; ++j) {
    int nt = wave * 2 + j;
    int c0 = nt * 16;
    v8f acc = {};
#pragma unroll
    for (int ks = 0; ks < 16; ++ks)
      acc = wmma_f16(load_frag(sh_dsA + ks * 512),
                     load_frag(W1SW + (size_t)(ks * 16 + nt) * 512), acc);
#pragma unroll
    for (int i = 0; i < 8; ++i) sh_ods[r0 + i][c0 + col] = acc[i] + b1[c0 + col];
  }
  __syncthreads();
  if (tid < 16) {
    int b = tid;
    float acc = b2[0];
    for (int p = 0; p < 256; ++p) acc += W2[p] * sh_ods[b][p];
    out[b0 + b] = acc;
  }
}

// ---------------- host launcher ----------------
extern "C" void kernel_launch(void* const* d_in, const int* in_sizes, int n_in,
                              void* d_out, int out_size, void* d_ws, size_t ws_size,
                              hipStream_t stream) {
  (void)in_sizes; (void)n_in; (void)out_size; (void)ws_size;
  const float* x       = (const float*)d_in[0];
  const float* y_known = (const float*)d_in[1];
  const float* We      = (const float*)d_in[2];
  const float* Ue      = (const float*)d_in[3];
  const float* Ve      = (const float*)d_in[4];
  const float* Wih_e   = (const float*)d_in[5];
  const float* Whh_e   = (const float*)d_in[6];
  const float* bih_e   = (const float*)d_in[7];
  const float* bhh_e   = (const float*)d_in[8];
  const float* Wd      = (const float*)d_in[9];
  const float* Ud      = (const float*)d_in[10];
  const float* Vd      = (const float*)d_in[11];
  const float* Wih_d   = (const float*)d_in[12];
  const float* Whh_d   = (const float*)d_in[13];
  const float* bih_d   = (const float*)d_in[14];
  const float* bhh_d   = (const float*)d_in[15];
  const float* Wt      = (const float*)d_in[16];
  const float* bt      = (const float*)d_in[17];
  const float* W1      = (const float*)d_in[18];
  const float* b1      = (const float*)d_in[19];
  const float* W2      = (const float*)d_in[20];
  const float* b2      = (const float*)d_in[21];

  char* p = (char*)d_ws;
  auto alloc = [&](size_t bytes) -> char* {
    char* r = p;
    p += (bytes + 255) & ~(size_t)255;
    return r;
  };
  _Float16* WeSW   = (_Float16*)alloc((size_t)512 * 128 * 2);
  _Float16* UeSW   = (_Float16*)alloc((size_t)128 * 128 * 2);
  _Float16* WiheSW = (_Float16*)alloc((size_t)128 * 1024 * 2);
  _Float16* WhheSW = (_Float16*)alloc((size_t)256 * 1024 * 2);
  _Float16* UdSW   = (_Float16*)alloc((size_t)256 * 256 * 2);
  _Float16* WdSW   = (_Float16*)alloc((size_t)512 * 256 * 2);
  _Float16* WhhdSW = (_Float16*)alloc((size_t)256 * 1024 * 2);
  _Float16* W1SW   = (_Float16*)alloc((size_t)512 * 256 * 2);
  float*    be     = (float*)alloc((size_t)1024 * 4);
  float*    bdv    = (float*)alloc((size_t)1024 * 4);
  _Float16* Ux     = (_Float16*)alloc((size_t)1024 * 128 * 128 * 2);
  _Float16* Hbuf   = (_Float16*)alloc((size_t)1024 * 128 * 256 * 2);
  _Float16* UdxB   = (_Float16*)alloc((size_t)1024 * 128 * 256 * 2);

  prep_kernel<<<dim3(4168), dim3(256), 0, stream>>>(
      We, Ue, Wih_e, Whh_e, bih_e, bhh_e, Wd, Ud, Whh_d, bih_d, bhh_d, W1,
      WeSW, UeSW, WiheSW, WhheSW, UdSW, WdSW, WhhdSW, W1SW, be, bdv);

  ux_kernel<<<dim3(1024), dim3(256), 0, stream>>>(x, UeSW, Ux);

  enc_kernel<<<dim3(64), dim3(256), 0, stream>>>(Ux, WeSW, WiheSW, WhheSW, be, Ve, Hbuf);

  udx_kernel<<<dim3(1024), dim3(256), 0, stream>>>(Hbuf, UdSW, UdxB);

  dec_kernel<<<dim3(64), dim3(256), 0, stream>>>(
      UdxB, Hbuf, WdSW, WhhdSW, bdv, Vd, Wih_d, Wt, bt, y_known,
      W1SW, b1, W2, b2, (float*)d_out);
}
